// GCN_Decoder_48155173322920
// MI455X (gfx1250) — matrix operand order repaired and checked by reading, not compiled, not run on hardware
//
#include <hip/hip_runtime.h>
#include <hip/hip_bf16.h>

typedef __attribute__((ext_vector_type(16))) _Float16 v16h;
typedef __attribute__((ext_vector_type(8)))  float    v8f;

#define N_NODES 100000
#define DFEAT   64
#define NTILES  (N_NODES / 16)
#define WPB     8            // waves per block (256 threads, wave32)
#define EPS_BN  1e-5f

// ---------- helpers ----------

// Guaranteed hardware f32 atomic add (no-return form), device scope.
__device__ __forceinline__ void atomAddF(float* p, float v) {
  asm volatile("global_atomic_add_f32 %0, %1, off scope:SCOPE_DEV"
               :: "v"(p), "v"(v) : "memory");
}

// Async copy of 16 bytes global -> LDS (ASYNCcnt-tracked, no VGPR data).
__device__ __forceinline__ void asyncCopy16(unsigned lds_off, const float* gaddr) {
  asm volatile("global_load_async_to_lds_b128 %0, %1, off"
               :: "v"(lds_off), "v"(gaddr) : "memory");
}

// B fragment (16-bit B 32x16, ISA 7.12.2): lane = column n (L&15),
// VGPR halves j hold K = 8*(L>>4) + j + (j>=8 ? 8 : 0).  W is row-major [64][64] f32.
__device__ __forceinline__ v16h load_b_frag(const float* __restrict__ W,
                                            int ct, int ks, int lane) {
  const int n = lane & 15, h = lane >> 4;
  v16h b;
#pragma unroll
  for (int j = 0; j < 16; ++j) {
    const int k = ks * 32 + 8 * h + j + (j >= 8 ? 8 : 0);
    b[j] = (_Float16)W[k * DFEAT + ct * 16 + n];
  }
  return b;
}

// A fragment from an f32 LDS tile (16x64 row-major).  16-bit A 16x32 layout
// (ISA 7.12.2): lane holds row m = L&15; half j -> K = base(j>>1)+8*(L>>4)+(j&1),
// base(v) = 2*(v&3) + 16*(v>>2).  Consecutive K pairs -> ds_load_b64 + cvt_pk.
// AFFINE fuses y = dv*x + bK[j] (GCN bias + dinv) before the f16 convert.
template <bool AFFINE>
__device__ __forceinline__ v16h make_a_frag(const float* ldsf, int ks, int lane,
                                            float dv, const float* bK) {
  const int m = lane & 15, h = lane >> 4;
  v16h a;
#pragma unroll
  for (int p = 0; p < 8; ++p) {
    const int kbase = ((p & 3) * 2 + (p >> 2) * 16) + 8 * h;
    const float2 xv = *(const float2*)&ldsf[m * DFEAT + ks * 32 + kbase];
    float x0 = xv.x, x1 = xv.y;
    if (AFFINE) {
      x0 = fmaf(x0, dv, bK[2 * p]);
      x1 = fmaf(x1, dv, bK[2 * p + 1]);
    }
    a[2 * p]     = (_Float16)x0;
    a[2 * p + 1] = (_Float16)x1;
  }
  return a;
}

// Stage one 16x64 f32 tile into this wave's LDS slice via async-to-LDS.
__device__ __forceinline__ void stage_tile_async(const float* __restrict__ src,
                                                 const float* ldsf, int lane) {
  // Previous iteration's ds_loads must have drained before overwriting LDS.
  asm volatile("s_wait_dscnt 0" ::: "memory");
  const unsigned ldsBase = (unsigned)(size_t)ldsf;   // addr[31:0] == LDS offset
#pragma unroll
  for (int i = lane; i < 256; i += 32)               // 256 x 16B = 4 KB tile
    asyncCopy16(ldsBase + i * 16, src + i * 4);
  asm volatile("s_wait_asynccnt 0" ::: "memory");
  __builtin_amdgcn_wave_barrier();
}

// ---------- small elementwise kernels ----------

__global__ void init_dinv_kernel(float* __restrict__ dinv) {
  int i = blockIdx.x * blockDim.x + threadIdx.x;
  if (i < N_NODES) dinv[i] = 1.0f;                   // self-loop degree
}

__global__ void count_deg_kernel(const int* __restrict__ ei2, int E,
                                 float* __restrict__ deg) {
  for (int e = blockIdx.x * blockDim.x + threadIdx.x; e < E;
       e += gridDim.x * blockDim.x) {
    const int d = ei2[2 * (E + e)];                  // low word of int64 dst
    atomAddF(&deg[d], 1.0f);
  }
}

__global__ void finalize_dinv_kernel(float* __restrict__ dinv) {
  int i = blockIdx.x * blockDim.x + threadIdx.x;
  if (i < N_NODES) dinv[i] = rsqrtf(dinv[i]);
}

__global__ void zero_stats_kernel(float* __restrict__ stats) {
  if (threadIdx.x < 128) stats[threadIdx.x] = 0.0f;
}

// agg[dst] += hp[src]; one edge per 16 lanes (float4 gather, 4 atomics/lane).
__global__ void scatter_kernel(const int* __restrict__ ei2, int E,
                               const float* __restrict__ hp,
                               float* __restrict__ agg) {
  const int total = E * 16;
  for (int idx = blockIdx.x * blockDim.x + threadIdx.x; idx < total;
       idx += gridDim.x * blockDim.x) {
    const int e  = idx >> 4;
    const int c4 = (idx & 15) << 2;
    __builtin_prefetch(&ei2[2 * (e + 4096)], 0, 1);  // global_prefetch_b8
    const int s = ei2[2 * e];
    const int d = ei2[2 * (E + e)];
    const float4 v = *(const float4*)&hp[(size_t)s * DFEAT + c4];
    float* dst = &agg[(size_t)d * DFEAT + c4];
    atomAddF(dst + 0, v.x);
    atomAddF(dst + 1, v.y);
    atomAddF(dst + 2, v.z);
    atomAddF(dst + 3, v.w);
  }
}

__global__ void bn_relu_kernel(const float* __restrict__ z,
                               const float* __restrict__ stats,
                               const float* __restrict__ g,
                               const float* __restrict__ bt,
                               float* __restrict__ out) {
  const int idx = blockIdx.x * blockDim.x + threadIdx.x;
  if (idx >= N_NODES * DFEAT) return;
  const int c = idx & 63;
  const float inv_n = 1.0f / (float)N_NODES;
  const float mu  = stats[c] * inv_n;
  const float var = fmaf(-mu, mu, stats[64 + c] * inv_n);
  const float y   = (z[idx] - mu) * rsqrtf(var + EPS_BN) * g[c] + bt[c];
  out[idx] = fmaxf(y, 0.0f);
}

// ---------- WMMA GEMM 1: hp = dinv ⊙ (X @ W); agg seeded with hp ----------

__global__ __launch_bounds__(256) void gemm_xw_kernel(
    const float* __restrict__ X, const float* __restrict__ W,
    const float* __restrict__ dinv, float* __restrict__ hp,
    float* __restrict__ agg) {
  __shared__ float ldsA[WPB][16 * DFEAT];            // f32 tiles, 4 KB/wave
  const int lane = threadIdx.x & 31;
  const int w    = threadIdx.x >> 5;
  const float* lds = ldsA[w];

  v16h Bf[2][4];
#pragma unroll
  for (int ks = 0; ks < 2; ++ks)
#pragma unroll
    for (int ct = 0; ct < 4; ++ct) Bf[ks][ct] = load_b_frag(W, ct, ks, lane);

  const int n = lane & 15, h = lane >> 4;

  for (int tile = blockIdx.x * WPB + w; tile < NTILES; tile += gridDim.x * WPB) {
    const int row0 = tile * 16;
    stage_tile_async(&X[(size_t)row0 * DFEAT], lds, lane);

    v8f acc[4] = {};
#pragma unroll
    for (int ks = 0; ks < 2; ++ks) {
      const v16h a = make_a_frag<false>(lds, ks, lane, 0.0f, nullptr);
#pragma unroll
      for (int ct = 0; ct < 4; ++ct)
        acc[ct] = __builtin_amdgcn_wmma_f32_16x16x32_f16(
            false, a, false, Bf[ks][ct], (short)0, acc[ct], false, false);
    }
    __builtin_amdgcn_wave_barrier();

#pragma unroll
    for (int jj = 0; jj < 8; ++jj) {
      const int r = row0 + jj + 8 * h;
      const float dv = dinv[r];
#pragma unroll
      for (int ct = 0; ct < 4; ++ct) {
        const float val = acc[ct][jj] * dv;
        hp [(size_t)r * DFEAT + ct * 16 + n] = val;
        agg[(size_t)r * DFEAT + ct * 16 + n] = val;  // self-loop seed
      }
    }
  }
}

// ---------- WMMA GEMM 2: z = (dinv ⊙ agg + b_gcn) @ fw + fb, BN stats ----------

__global__ __launch_bounds__(256) void gemm_fc_kernel(
    const float* __restrict__ agg, const float* __restrict__ dinv,
    const float* __restrict__ bgcn, const float* __restrict__ fw,
    const float* __restrict__ fb, float* __restrict__ z,
    float* __restrict__ stats) {
  __shared__ float ldsA[WPB][16 * DFEAT];
  const int lane = threadIdx.x & 31;
  const int w    = threadIdx.x >> 5;
  const float* lds = ldsA[w];

  v16h Bf[2][4];
#pragma unroll
  for (int ks = 0; ks < 2; ++ks)
#pragma unroll
    for (int ct = 0; ct < 4; ++ct) Bf[ks][ct] = load_b_frag(fw, ct, ks, lane);

  const int n = lane & 15, h = lane >> 4;

  // Per-lane GCN bias replicated into A-fragment K-order (loop invariant).
  float bK[2][16];
#pragma unroll
  for (int ks = 0; ks < 2; ++ks)
#pragma unroll
    for (int p = 0; p < 8; ++p) {
      const int kbase = ((p & 3) * 2 + (p >> 2) * 16) + 8 * h;
      bK[ks][2 * p]     = bgcn[ks * 32 + kbase];
      bK[ks][2 * p + 1] = bgcn[ks * 32 + kbase + 1];
    }

  float fbv[4];
#pragma unroll
  for (int ct = 0; ct < 4; ++ct) fbv[ct] = fb[ct * 16 + n];

  for (int tile = blockIdx.x * WPB + w; tile < NTILES; tile += gridDim.x * WPB) {
    const int row0 = tile * 16;
    stage_tile_async(&agg[(size_t)row0 * DFEAT], lds, lane);
    const float dv = dinv[row0 + (lane & 15)];       // this lane's A row

    v8f acc[4] = {};
#pragma unroll
    for (int ks = 0; ks < 2; ++ks) {
      const v16h a = make_a_frag<true>(lds, ks, lane, dv, bK[ks]);
#pragma unroll
      for (int ct = 0; ct < 4; ++ct)
        acc[ct] = __builtin_amdgcn_wmma_f32_16x16x32_f16(
            false, a, false, Bf[ks][ct], (short)0, acc[ct], false, false);
    }
    __builtin_amdgcn_wave_barrier();

#pragma unroll
    for (int ct = 0; ct < 4; ++ct) {
      const int col = ct * 16 + n;
      float s = 0.0f, s2 = 0.0f;
#pragma unroll
      for (int jj = 0; jj < 8; ++jj) {
        const int r = row0 + jj + 8 * h;
        const float zv = acc[ct][jj] + fbv[ct];
        z[(size_t)r * DFEAT + col] = zv;
        s  += zv;
        s2 += zv * zv;
      }
      atomAddF(&stats[col], s);        // BN sum
      atomAddF(&stats[64 + col], s2);  // BN sum of squares
    }
  }
}

// ---------- launch ----------

extern "C" void kernel_launch(void* const* d_in, const int* in_sizes, int n_in,
                              void* d_out, int out_size, void* d_ws, size_t ws_size,
                              hipStream_t stream) {
  const float* x   = (const float*)d_in[0];
  const int*   ei2 = (const int*)d_in[1];   // int64 edge_index as int pairs (LE low word)
  const float* W1  = (const float*)d_in[2];
  const float* b1  = (const float*)d_in[3];
  const float* fw1 = (const float*)d_in[4];
  const float* fb1 = (const float*)d_in[5];
  const float* g1  = (const float*)d_in[6];
  const float* bt1 = (const float*)d_in[7];
  const float* W2  = (const float*)d_in[8];
  const float* b2  = (const float*)d_in[9];
  const float* fw2 = (const float*)d_in[10];
  const float* fb2 = (const float*)d_in[11];
  const float* g2  = (const float*)d_in[12];
  const float* bt2 = (const float*)d_in[13];
  const int E = in_sizes[1] / 2;

  float* ws    = (float*)d_ws;
  float* dinv  = ws;                                   // N
  float* hp    = ws + N_NODES;                         // N*64
  float* agg   = hp   + (size_t)N_NODES * DFEAT;       // N*64
  float* zbuf  = agg  + (size_t)N_NODES * DFEAT;       // N*64
  float* tbuf  = zbuf + (size_t)N_NODES * DFEAT;       // N*64 (layer-1 output)
  float* stats = tbuf + (size_t)N_NODES * DFEAT;       // 128
  float* outp  = (float*)d_out;

  const int TPB     = 256;
  const int gElemN  = (N_NODES + TPB - 1) / TPB;
  const int gElemND = (N_NODES * DFEAT + TPB - 1) / TPB;
  const int gGemm   = (NTILES + WPB - 1) / WPB;

  init_dinv_kernel<<<gElemN, TPB, 0, stream>>>(dinv);
  count_deg_kernel<<<4096, TPB, 0, stream>>>(ei2, E, dinv);
  finalize_dinv_kernel<<<gElemN, TPB, 0, stream>>>(dinv);

  // ----- layer 1 -----
  zero_stats_kernel<<<1, 128, 0, stream>>>(stats);
  gemm_xw_kernel<<<gGemm, TPB, 0, stream>>>(x, W1, dinv, hp, agg);
  scatter_kernel<<<8192, TPB, 0, stream>>>(ei2, E, hp, agg);
  gemm_fc_kernel<<<gGemm, TPB, 0, stream>>>(agg, dinv, b1, fw1, fb1, zbuf, stats);
  bn_relu_kernel<<<gElemND, TPB, 0, stream>>>(zbuf, stats, g1, bt1, tbuf);

  // ----- layer 2 -----
  zero_stats_kernel<<<1, 128, 0, stream>>>(stats);
  gemm_xw_kernel<<<gGemm, TPB, 0, stream>>>(tbuf, W2, dinv, hp, agg);
  scatter_kernel<<<8192, TPB, 0, stream>>>(ei2, E, hp, agg);
  gemm_fc_kernel<<<gGemm, TPB, 0, stream>>>(agg, dinv, b2, fw2, fb2, zbuf, stats);
  bn_relu_kernel<<<gElemND, TPB, 0, stream>>>(zbuf, stats, g2, bt2, outp);
}
